// Attention_57166014709861
// MI455X (gfx1250) — matrix operand order repaired and checked by reading, not compiled, not run on hardware
//
#include <hip/hip_runtime.h>
#include <hip/hip_bf16.h>

typedef __bf16 bf16;
typedef __attribute__((ext_vector_type(8)))  __bf16 v8bf;
typedef __attribute__((ext_vector_type(16))) __bf16 v16bf;
typedef __attribute__((ext_vector_type(8)))  float  v8f;

#define NSEQ 1024
#define DHEAD 48
#define CDIM 384

__device__ inline v16bf cat8(v8bf lo, v8bf hi) {
  return __builtin_shufflevector(lo, hi, 0,1,2,3,4,5,6,7,8,9,10,11,12,13,14,15);
}
__device__ inline v8bf  ld8 (const bf16* p) { return *reinterpret_cast<const v8bf*>(p); }
__device__ inline v16bf ld16(const bf16* p) { return *reinterpret_cast<const v16bf*>(p); }

__device__ inline v8f wmma_bf16(v16bf a, v16bf b, v8f c) {
  // D = A(16x32 bf16) * B(32x16 bf16) + C(16x16 f32)
  return __builtin_amdgcn_wmma_f32_16x16x32_bf16(false, a, false, b, (short)0, c, false, false);
}

// ---------------- prep: weights fp32 -> bf16 ----------------
__global__ __launch_bounds__(256) void convert_w_kernel(const float* __restrict__ wq,
                                                        const float* __restrict__ wp,
                                                        bf16* __restrict__ wq_bf,
                                                        bf16* __restrict__ wp_bf) {
  int i = blockIdx.x * 256 + threadIdx.x;
  if (i < 3 * CDIM * CDIM) wq_bf[i] = (bf16)wq[i];
  if (i < CDIM * CDIM)     wp_bf[i] = (bf16)wp[i];
}

// ---------------- prep: x[b][c][n] f32 -> xt[b][n][c] bf16 ----------------
__global__ __launch_bounds__(256) void transpose_x_kernel(const float* __restrict__ x,
                                                          bf16* __restrict__ xt) {
  __shared__ float tile[32][33];
  int b  = blockIdx.z;
  int n0 = blockIdx.x * 32, c0 = blockIdx.y * 32;
  int tx = threadIdx.x, ty = threadIdx.y;
#pragma unroll
  for (int i = ty; i < 32; i += 8)
    tile[i][tx] = x[((size_t)b * CDIM + c0 + i) * NSEQ + n0 + tx];
  __syncthreads();
#pragma unroll
  for (int i = ty; i < 32; i += 8)
    xt[((size_t)b * NSEQ + n0 + i) * CDIM + c0 + tx] = (bf16)tile[tx][i];
}

// ---------------- QKV GEMM: wave computes 16(o) x 64(n) strip, 4 indep accumulators ----
// Scatters into Q[b][h][n][48] (scaled by 48^-.5 * log2(e)), K[b][h][n][48], Vt[b][h][d][n].
__global__ __launch_bounds__(256) void qkv_kernel(const bf16* __restrict__ W,
                                                  const bf16* __restrict__ Xt,
                                                  bf16* __restrict__ Q,
                                                  bf16* __restrict__ K,
                                                  bf16* __restrict__ Vt) {
  int b    = blockIdx.y;
  int w    = threadIdx.x >> 5;
  int lane = threadIdx.x & 31;
  int g    = lane >> 4, ln = lane & 15;
  int tl   = blockIdx.x * 8 + w;   // 0..1151
  int ot   = tl % 72, ns = tl / 72;

  const bf16* arow = W + (size_t)(ot * 16 + ln) * CDIM;
  const bf16* brow = Xt + ((size_t)b * NSEQ + ns * 64 + ln) * CDIM;

  v8f acc0 = {}, acc1 = {}, acc2 = {}, acc3 = {};
  for (int kb = 0; kb < CDIM; kb += 32) {
    v16bf a = ld16(arow + kb + g * 16);
    acc0 = wmma_bf16(a, ld16(brow + (size_t)(0 * 16) * CDIM + kb + g * 16), acc0);
    acc1 = wmma_bf16(a, ld16(brow + (size_t)(1 * 16) * CDIM + kb + g * 16), acc1);
    acc2 = wmma_bf16(a, ld16(brow + (size_t)(2 * 16) * CDIM + kb + g * 16), acc2);
    acc3 = wmma_bf16(a, ld16(brow + (size_t)(3 * 16) * CDIM + kb + g * 16), acc3);
  }

  const float qscale = 0.14433756729740643f * 1.4426950408889634f;  // 48^-0.5 * log2(e)
#pragma unroll
  for (int t = 0; t < 4; ++t) {
    v8f acc = (t == 0) ? acc0 : (t == 1) ? acc1 : (t == 2) ? acc2 : acc3;
    int n = ns * 64 + t * 16 + ln;
#pragma unroll
    for (int r = 0; r < 8; ++r) {
      int o = ot * 16 + r + g * 8;
      float v = acc[r];
      if (o < CDIM) {
        int h = o / 48, dd = o % 48;
        Q[(((size_t)b * 8 + h) * NSEQ + n) * DHEAD + dd] = (bf16)(v * qscale);
      } else if (o < 2 * CDIM) {
        int oc = o - CDIM; int h = oc / 48, dd = oc % 48;
        K[(((size_t)b * 8 + h) * NSEQ + n) * DHEAD + dd] = (bf16)v;
      } else {
        int oc = o - 2 * CDIM; int h = oc / 48, dd = oc % 48;
        Vt[(((size_t)b * 8 + h) * DHEAD + dd) * NSEQ + n] = (bf16)v;
      }
    }
  }
}

// ---------------- Flash attention: WG per (b,h,128 q rows); wave = 16 q rows, 64 keys/iter
__global__ __launch_bounds__(256) void attn_kernel(const bf16* __restrict__ Q,
                                                   const bf16* __restrict__ K,
                                                   const bf16* __restrict__ Vt,
                                                   bf16* __restrict__ Ot) {
  __shared__ __align__(32) bf16 plds[8][16][64];   // per-wave 16x64 P tile

  int idx = blockIdx.x;
  int b = idx >> 6, h = (idx >> 3) & 7, qb = idx & 7;
  int w    = threadIdx.x >> 5;
  int lane = threadIdx.x & 31;
  int g    = lane >> 4, ln = lane & 15;
  int bh   = b * 8 + h;
  int q0   = qb * 128 + w * 16;

  // Q A-fragments (16x64, K=48..63 zeroed in registers)
  const bf16* qrow = Q + ((size_t)bh * NSEQ + (q0 + ln)) * DHEAD;
  v16bf a0 = ld16(qrow + g * 16);
  v8bf  z8 = {};
  v16bf a1 = {};
  if (g == 0) a1 = ld16(qrow + 32);

  const bf16* kbase = K  + (size_t)bh * NSEQ * DHEAD;
  const bf16* vbase = Vt + (size_t)bh * DHEAD * NSEQ;

  float m[8], l[8];
  v8f o0 = {}, o1 = {}, o2 = {};
#pragma unroll
  for (int r = 0; r < 8; ++r) { m[r] = -3.0e38f; l[r] = 0.f; }

  for (int j = 0; j < 16; ++j) {
    int y0 = j * 64;
    if (j < 15)  // prefetch next K chunk (global_prefetch_b8)
      __builtin_prefetch(kbase + (size_t)(y0 + 64 + lane * 4) * DHEAD, 0, 0);

    // ---- S = Q*K^T for 4 key tiles (keys y0..y0+63)
    v8f s0 = {}, s1 = {}, s2 = {}, s3 = {};
#pragma unroll
    for (int t = 0; t < 4; ++t) {
      const bf16* krow = kbase + (size_t)(y0 + t * 16 + ln) * DHEAD;
      v16bf bf0 = ld16(krow + g * 16);
      v16bf bf1 = {};
      if (g == 0) bf1 = ld16(krow + 32);   // K=32..47; high lane group stays zero (d pad)
      v8f s = (t == 0) ? s0 : (t == 1) ? s1 : (t == 2) ? s2 : s3;
      s = wmma_bf16(a0, bf0, s);
      s = wmma_bf16(a1, bf1, s);
      if (t == 0) s0 = s; else if (t == 1) s1 = s; else if (t == 2) s2 = s; else s3 = s;
    }

    // ---- online softmax (base-2; log2(e) folded into Q scale)
    float alpha[8];
#pragma unroll
    for (int r = 0; r < 8; ++r) {
      float v = fmaxf(fmaxf(s0[r], s1[r]), fmaxf(s2[r], s3[r]));
      v = fmaxf(v, __shfl_xor(v, 8));
      v = fmaxf(v, __shfl_xor(v, 4));
      v = fmaxf(v, __shfl_xor(v, 2));
      v = fmaxf(v, __shfl_xor(v, 1));
      float mn = fmaxf(m[r], v);
      alpha[r] = exp2f(m[r] - mn);
      m[r] = mn;
    }
#pragma unroll
    for (int r = 0; r < 8; ++r) {
      float p0 = exp2f(s0[r] - m[r]);
      float p1 = exp2f(s1[r] - m[r]);
      float p2 = exp2f(s2[r] - m[r]);
      float p3 = exp2f(s3[r] - m[r]);
      float rs = (p0 + p1) + (p2 + p3);
      rs += __shfl_xor(rs, 8);
      rs += __shfl_xor(rs, 4);
      rs += __shfl_xor(rs, 2);
      rs += __shfl_xor(rs, 1);
      l[r] = l[r] * alpha[r] + rs;
      o0[r] *= alpha[r]; o1[r] *= alpha[r]; o2[r] *= alpha[r];
      // stage P tile (row-major 16x64 bf16) for A-fragment reload
      plds[w][r + g * 8][ln]      = (bf16)p0;
      plds[w][r + g * 8][16 + ln] = (bf16)p1;
      plds[w][r + g * 8][32 + ln] = (bf16)p2;
      plds[w][r + g * 8][48 + ln] = (bf16)p3;
    }
    asm volatile("s_wait_dscnt 0" ::: "memory");  // per-wave LDS store->load ordering
    const bf16* pr = &plds[w][ln][0];
    v16bf pa0 = *(const v16bf*)(pr + g * 16);
    v16bf pa1 = *(const v16bf*)(pr + 32 + g * 16);

    // ---- O += P*V  (3 output d-tiles x 2 K-chunks)
#pragma unroll
    for (int dt = 0; dt < 3; ++dt) {
      const bf16* vrow = vbase + (size_t)(dt * 16 + ln) * NSEQ + y0;
      v16bf vb0 = ld16(vrow + g * 16);
      v16bf vb1 = ld16(vrow + 32 + g * 16);
      v8f o = (dt == 0) ? o0 : (dt == 1) ? o1 : o2;
      o = wmma_bf16(pa0, vb0, o);
      o = wmma_bf16(pa1, vb1, o);
      if (dt == 0) o0 = o; else if (dt == 1) o1 = o; else o2 = o;
    }
  }

  // ---- normalize & store heads-concatenated [b][n][384]
#pragma unroll
  for (int r = 0; r < 8; ++r) {
    float inv = 1.0f / l[r];
    int n = q0 + r + g * 8;
    bf16* dst = Ot + ((size_t)b * NSEQ + n) * CDIM + h * 48 + ln;
    dst[0]  = (bf16)(o0[r] * inv);
    dst[16] = (bf16)(o1[r] * inv);
    dst[32] = (bf16)(o2[r] * inv);
  }
}

// ---------------- Proj GEMM: wave computes 16(o) x 64(n) strip + bias, f32 out ----------
__global__ __launch_bounds__(256) void proj_kernel(const bf16* __restrict__ Wp,
                                                   const bf16* __restrict__ At,
                                                   const float* __restrict__ bias,
                                                   float* __restrict__ out) {
  int b    = blockIdx.y;
  int w    = threadIdx.x >> 5;
  int lane = threadIdx.x & 31;
  int g    = lane >> 4, ln = lane & 15;
  int tl   = blockIdx.x * 8 + w;   // 0..383
  int ot   = tl % 24, ns = tl / 24;

  const bf16* arow = Wp + (size_t)(ot * 16 + ln) * CDIM;
  const bf16* brow = At + ((size_t)b * NSEQ + ns * 64 + ln) * CDIM;

  v8f acc0 = {}, acc1 = {}, acc2 = {}, acc3 = {};
  for (int kb = 0; kb < CDIM; kb += 32) {
    v16bf a = ld16(arow + kb + g * 16);
    acc0 = wmma_bf16(a, ld16(brow + (size_t)(0 * 16) * CDIM + kb + g * 16), acc0);
    acc1 = wmma_bf16(a, ld16(brow + (size_t)(1 * 16) * CDIM + kb + g * 16), acc1);
    acc2 = wmma_bf16(a, ld16(brow + (size_t)(2 * 16) * CDIM + kb + g * 16), acc2);
    acc3 = wmma_bf16(a, ld16(brow + (size_t)(3 * 16) * CDIM + kb + g * 16), acc3);
  }
#pragma unroll
  for (int t = 0; t < 4; ++t) {
    v8f acc = (t == 0) ? acc0 : (t == 1) ? acc1 : (t == 2) ? acc2 : acc3;
#pragma unroll
    for (int r = 0; r < 8; ++r) {
      int o = ot * 16 + r + g * 8;
      int n = ns * 64 + t * 16 + ln;
      out[((size_t)b * CDIM + o) * NSEQ + n] = acc[r] + bias[o];
    }
  }
}

extern "C" void kernel_launch(void* const* d_in, const int* in_sizes, int n_in,
                              void* d_out, int out_size, void* d_ws, size_t ws_size,
                              hipStream_t stream) {
  (void)in_sizes; (void)n_in; (void)out_size; (void)ws_size;
  const float* x      = (const float*)d_in[0];
  const float* w_qkv  = (const float*)d_in[1];
  const float* w_proj = (const float*)d_in[2];
  const float* b_proj = (const float*)d_in[3];
  float* out = (float*)d_out;

  char* ws = (char*)d_ws;
  bf16* wq_bf = (bf16*)(ws + 0);          //  1152*384*2 = 884736
  bf16* wp_bf = (bf16*)(ws + 884736);     //   384*384*2 = 294912
  bf16* xt    = (bf16*)(ws + 1179648);    // 8*1024*384*2 = 6291456
  bf16* Qb    = (bf16*)(ws + 7471104);    // 8*8*1024*48*2
  bf16* Kb    = (bf16*)(ws + 13762560);   // 8*8*1024*48*2
  bf16* Vt    = (bf16*)(ws + 20054016);   // 8*8*48*1024*2
  bf16* At    = (bf16*)(ws + 26345472);   // 8*1024*384*2   (total ~31.1 MB)

  convert_w_kernel<<<1728, 256, 0, stream>>>(w_qkv, w_proj, wq_bf, wp_bf);
  transpose_x_kernel<<<dim3(32, 12, 8), dim3(32, 8), 0, stream>>>(x, xt);
  qkv_kernel<<<dim3(144, 8), 256, 0, stream>>>(wq_bf, xt, Qb, Kb, Vt);
  attn_kernel<<<512, 256, 0, stream>>>(Qb, Kb, Vt, At);
  proj_kernel<<<dim3(48, 8), 256, 0, stream>>>(wp_bf, At, b_proj, out);
}
